// multi_head_attention_layer_76940044140881
// MI455X (gfx1250) — matrix operand order repaired and checked by reading, not compiled, not run on hardware
//
#include <hip/hip_runtime.h>

typedef __attribute__((ext_vector_type(16))) __bf16 v16bf;
typedef __attribute__((ext_vector_type(8)))  __bf16 v8bf;
typedef __attribute__((ext_vector_type(8)))  float  v8f;

#define WMMA_BF16(a, b, c) \
    __builtin_amdgcn_wmma_f32_16x16x32_bf16(false, (a), false, (b), (short)0, (c), false, false)

// ---------------------------------------------------------------------------
// Weight convert + transpose, LDS-tiled (coalesced read AND write):
// W (1024 x 1024, f32) -> Wt (1024 x 1024 transposed, bf16)
// ---------------------------------------------------------------------------
__global__ __launch_bounds__(256) void wconv_kernel(const float* __restrict__ W,
                                                    __bf16* __restrict__ Wt) {
    __shared__ float tile[32][33];
    const int tx = threadIdx.x;        // 0..31
    const int ty = threadIdx.y;        // 0..7
    const int i0 = blockIdx.y * 32;    // input-row block
    const int o0 = blockIdx.x * 32;    // output-col block
#pragma unroll
    for (int r = 0; r < 32; r += 8)
        tile[ty + r][tx] = W[(size_t)(i0 + ty + r) * 1024 + o0 + tx];
    __syncthreads();
#pragma unroll
    for (int r = 0; r < 32; r += 8)
        Wt[(size_t)(o0 + ty + r) * 1024 + i0 + tx] = (__bf16)tile[tx][ty + r];
}

// ---------------------------------------------------------------------------
// Projection GEMM: C = X(f32, Mx1024) @ W, W transposed bf16 (1024x1024).
// One wave computes a 32x64 tile (2 row-fragments share every B fragment).
// MODE 0: store [b][h][s][dh] (Q,K);  MODE 1: store [b][h][dh][s] (V^T)
// ---------------------------------------------------------------------------
template <int MODE>
__global__ __launch_bounds__(32) void gemm_proj(const float* __restrict__ X,
                                                const __bf16* __restrict__ Wt,
                                                __bf16* __restrict__ dst) {
    const int lane = threadIdx.x;
    const int lm = lane & 15;
    const int hi = lane >> 4;
    const int m0 = blockIdx.x * 32;
    const int n0 = blockIdx.y * 64;

    v8f acc[2][4] = {};
    const float* xrow0 = X + (size_t)(m0 + lm) * 1024;
    const float* xrow1 = X + (size_t)(m0 + 16 + lm) * 1024;

    for (int kb = 0; kb < 1024; kb += 32) {
        v16bf aF[2];
        {
            v8f alo = *(const v8f*)(xrow0 + kb + hi * 16);
            v8f ahi = *(const v8f*)(xrow0 + kb + 8 + hi * 16);
            v8f blo = *(const v8f*)(xrow1 + kb + hi * 16);
            v8f bhi = *(const v8f*)(xrow1 + kb + 8 + hi * 16);
#pragma unroll
            for (int i = 0; i < 8; ++i) {
                aF[0][i] = (__bf16)alo[i]; aF[0][8 + i] = (__bf16)ahi[i];
                aF[1][i] = (__bf16)blo[i]; aF[1][8 + i] = (__bf16)bhi[i];
            }
        }
#pragma unroll
        for (int j = 0; j < 4; ++j) {
            const __bf16* bp = Wt + (size_t)(n0 + j * 16 + lm) * 1024 + kb + hi * 16;
            v16bf bF = *(const v16bf*)bp;                 // shared by both row groups
            acc[0][j] = WMMA_BF16(aF[0], bF, acc[0][j]);
            acc[1][j] = WMMA_BF16(aF[1], bF, acc[1][j]);
        }
    }

#pragma unroll
    for (int rg = 0; rg < 2; ++rg)
#pragma unroll
        for (int j = 0; j < 4; ++j)
#pragma unroll
            for (int g = 0; g < 8; ++g) {
                int r = m0 + rg * 16 + g + hi * 8;   // row = (b,s)
                int c = n0 + j * 16 + lm;            // col = (h,dh)
                int b = r >> 11, s = r & 2047, h = c >> 6, dh = c & 63;
                size_t idx;
                if (MODE == 0) idx = (((size_t)b * 16 + h) * 2048 + s) * 64 + dh;
                else           idx = (((size_t)b * 16 + h) * 64 + dh) * 2048 + s;
                dst[idx] = (__bf16)acc[rg][j][g];
            }
}

// ---------------------------------------------------------------------------
// Streaming attention, one wave = (b, h, 32-row query tile).
// Linear row-sum normalization: num += (QK^T/8)V, den += (QK^T/8)*1
// K/V B-fragments are shared across the two query row groups.
// ---------------------------------------------------------------------------
__global__ __launch_bounds__(32) void attn_kernel(const __bf16* __restrict__ Q,
                                                  const __bf16* __restrict__ K,
                                                  const __bf16* __restrict__ Vt,
                                                  __bf16* __restrict__ att) {
    __shared__ __bf16 Pl[32 * 32];          // per-wave P-tile scratch (1 wave/block)

    const int lane = threadIdx.x;
    const int lm = lane & 15;
    const int hi = lane >> 4;
    const int q0 = blockIdx.x * 32;
    const int h  = blockIdx.y;
    const int b  = blockIdx.z;
    const size_t bh = (size_t)b * 16 + h;

    const __bf16* qp = Q + (bh * 2048 + q0) * 64;
    const __bf16* kp = K + bh * 2048 * 64;
    const __bf16* vp = Vt + bh * 64 * 2048;

    // Q A-fragments: [row group][dh slice]
    v16bf qa[2][2];
#pragma unroll
    for (int rg = 0; rg < 2; ++rg)
#pragma unroll
        for (int s2 = 0; s2 < 2; ++s2) {
            const __bf16* r = qp + (size_t)(rg * 16 + lm) * 64 + 32 * s2 + hi * 16;
            v8bf lo = *(const v8bf*)r;
            v8bf hh = *(const v8bf*)(r + 8);
#pragma unroll
            for (int i = 0; i < 8; ++i) { qa[rg][s2][i] = lo[i]; qa[rg][s2][8 + i] = hh[i]; }
        }

    // ones B-fragment: column 0 -> row sums
    v16bf ones;
    {
        __bf16 o1 = (lm == 0) ? (__bf16)1.0f : (__bf16)0.0f;
#pragma unroll
        for (int i = 0; i < 16; ++i) ones[i] = o1;
    }

    v8f accN[2][4] = {};
    v8f accD[2] = {};

    for (int kt = 0; kt < 2048; kt += 32) {
        // ---- scores: S = Q(32x64) @ K_tile^T(64x32)
        v8f sc[2][2] = {};
#pragma unroll
        for (int t = 0; t < 2; ++t)
#pragma unroll
            for (int s2 = 0; s2 < 2; ++s2) {
                const __bf16* krow = kp + (size_t)(kt + t * 16 + lm) * 64 + 32 * s2 + hi * 16;
                v16bf kF = *(const v16bf*)krow;           // shared by both row groups
                sc[0][t] = WMMA_BF16(qa[0][s2], kF, sc[0][t]);
                sc[1][t] = WMMA_BF16(qa[1][s2], kF, sc[1][t]);
            }
        // ---- scale by 1/sqrt(64); LDS round-trip re-layouts C->A
#pragma unroll
        for (int rg = 0; rg < 2; ++rg)
#pragma unroll
            for (int t = 0; t < 2; ++t)
#pragma unroll
                for (int g = 0; g < 8; ++g)
                    Pl[(rg * 16 + g + hi * 8) * 32 + t * 16 + lm] =
                        (__bf16)(sc[rg][t][g] * 0.125f);

        v16bf pa[2];
#pragma unroll
        for (int rg = 0; rg < 2; ++rg) {
            const __bf16* pr = Pl + (rg * 16 + lm) * 32 + hi * 16;
            v8bf lo = *(const v8bf*)pr;
            v8bf hh = *(const v8bf*)(pr + 8);
#pragma unroll
            for (int i = 0; i < 8; ++i) { pa[rg][i] = lo[i]; pa[rg][8 + i] = hh[i]; }
        }
        // ---- num += P(32x32) @ V_tile(32x64), den += P @ ones
#pragma unroll
        for (int j = 0; j < 4; ++j) {
            const __bf16* vrow = vp + (size_t)(j * 16 + lm) * 2048 + kt + hi * 16;
            v16bf vF = *(const v16bf*)vrow;               // shared by both row groups
            accN[0][j] = WMMA_BF16(pa[0], vF, accN[0][j]);
            accN[1][j] = WMMA_BF16(pa[1], vF, accN[1][j]);
        }
        accD[0] = WMMA_BF16(pa[0], ones, accD[0]);
        accD[1] = WMMA_BF16(pa[1], ones, accD[1]);
    }

    // ---- divide rows by (rowsum + 1e-9), store att[b][s][h*64+dh] (bf16)
#pragma unroll
    for (int rg = 0; rg < 2; ++rg)
#pragma unroll
        for (int g = 0; g < 8; ++g) {
            float d   = accD[rg][g];
            float dlo = __shfl(d, 0, 32);    // den for row rg*16+g
            float dhi = __shfl(d, 16, 32);   // den for row rg*16+g+8
            float den = ((lane < 16) ? dlo : dhi) + 1e-9f;
            float rd  = 1.0f / den;
            int r = q0 + rg * 16 + g + hi * 8;
            size_t rowbase = ((size_t)b * 2048 + r) * 1024 + h * 64;
#pragma unroll
            for (int j = 0; j < 4; ++j)
                att[rowbase + j * 16 + lm] = (__bf16)(accN[rg][j][g] * rd);
        }
}

// ---------------------------------------------------------------------------
// Output GEMM: out(f32) = att(bf16, Mx1024) @ Wo_t, 32x64 tile per wave
// ---------------------------------------------------------------------------
__global__ __launch_bounds__(32) void gemm_out(const __bf16* __restrict__ A,
                                               const __bf16* __restrict__ Wt,
                                               float* __restrict__ out) {
    const int lane = threadIdx.x;
    const int lm = lane & 15;
    const int hi = lane >> 4;
    const int m0 = blockIdx.x * 32;
    const int n0 = blockIdx.y * 64;

    v8f acc[2][4] = {};
    const __bf16* arow0 = A + (size_t)(m0 + lm) * 1024;
    const __bf16* arow1 = A + (size_t)(m0 + 16 + lm) * 1024;

    for (int kb = 0; kb < 1024; kb += 32) {
        v16bf aF[2];
        {
            v8bf lo0 = *(const v8bf*)(arow0 + kb + hi * 16);
            v8bf hh0 = *(const v8bf*)(arow0 + kb + 8 + hi * 16);
            v8bf lo1 = *(const v8bf*)(arow1 + kb + hi * 16);
            v8bf hh1 = *(const v8bf*)(arow1 + kb + 8 + hi * 16);
#pragma unroll
            for (int i = 0; i < 8; ++i) {
                aF[0][i] = lo0[i]; aF[0][8 + i] = hh0[i];
                aF[1][i] = lo1[i]; aF[1][8 + i] = hh1[i];
            }
        }
#pragma unroll
        for (int j = 0; j < 4; ++j) {
            const __bf16* bp = Wt + (size_t)(n0 + j * 16 + lm) * 1024 + kb + hi * 16;
            v16bf bF = *(const v16bf*)bp;
            acc[0][j] = WMMA_BF16(aF[0], bF, acc[0][j]);
            acc[1][j] = WMMA_BF16(aF[1], bF, acc[1][j]);
        }
    }

#pragma unroll
    for (int rg = 0; rg < 2; ++rg)
#pragma unroll
        for (int j = 0; j < 4; ++j)
#pragma unroll
            for (int g = 0; g < 8; ++g) {
                int r = m0 + rg * 16 + g + hi * 8;
                int c = n0 + j * 16 + lm;
                out[(size_t)r * 1024 + c] = acc[rg][j][g];
            }
}

// ---------------------------------------------------------------------------
extern "C" void kernel_launch(void* const* d_in, const int* in_sizes, int n_in,
                              void* d_out, int out_size, void* d_ws, size_t ws_size,
                              hipStream_t stream) {
    const float* q  = (const float*)d_in[0];
    const float* k  = (const float*)d_in[1];
    const float* v  = (const float*)d_in[2];
    const float* Wq = (const float*)d_in[3];
    const float* Wk = (const float*)d_in[4];
    const float* Wv = (const float*)d_in[5];
    const float* Wo = (const float*)d_in[6];

    __bf16* ws = (__bf16*)d_ws;
    const size_t WT = 1024ull * 1024ull;   // elements per weight matrix
    __bf16* wq_t = ws + 0 * WT;
    __bf16* wk_t = ws + 1 * WT;
    __bf16* wv_t = ws + 2 * WT;
    __bf16* wo_t = ws + 3 * WT;
    __bf16* qb   = ws + 4 * WT;    // [B][H][S][64]   8M elems
    __bf16* kb   = ws + 12 * WT;   // [B][H][S][64]
    __bf16* vt   = ws + 20 * WT;   // [B][H][64][S]
    __bf16* att  = ws + 28 * WT;   // [B][S][1024]    (total 72 MB of ws)

    // 1) weights -> transposed bf16 (LDS-tiled, fully coalesced)
    dim3 wgrid(32, 32), wblk(32, 8);
    wconv_kernel<<<wgrid, wblk, 0, stream>>>(Wq, wq_t);
    wconv_kernel<<<wgrid, wblk, 0, stream>>>(Wk, wk_t);
    wconv_kernel<<<wgrid, wblk, 0, stream>>>(Wv, wv_t);
    wconv_kernel<<<wgrid, wblk, 0, stream>>>(Wo, wo_t);

    // 2) projections (one wave = 32x64 tile)
    dim3 pgrid(256, 16);
    gemm_proj<0><<<pgrid, 32, 0, stream>>>(q, wq_t, qb);
    gemm_proj<0><<<pgrid, 32, 0, stream>>>(k, wk_t, kb);
    gemm_proj<1><<<pgrid, 32, 0, stream>>>(v, wv_t, vt);

    // 3) streaming attention with linear row-sum normalization
    dim3 agrid(64, 16, 4);   // (S/32, H, B)
    attn_kernel<<<agrid, 32, 0, stream>>>(qb, kb, vt, att);

    // 4) output projection -> f32
    gemm_out<<<pgrid, 32, 0, stream>>>(att, wo_t, (float*)d_out);
}
